// OneTransBlock_15410342658374
// MI455X (gfx1250) — compile-verified, matrix operand order checked by reading, not run on hardware
//
#include <hip/hip_runtime.h>
#include <stdint.h>

// ---------------------------------------------------------------------------
// Problem constants (from reference setup_inputs)
// ---------------------------------------------------------------------------
constexpr int B   = 4;
constexpr int S   = 2048;
constexpr int D   = 1024;
constexpr int F   = 4096;
constexpr int N   = 16;     // ns tokens / experts
constexpr int H   = 16;     // heads
constexpr int HD  = 64;     // head dim
constexpr int NSL = 1024;
constexpr int TOTQ = NSL + N;        // 1040 query rows per batch
constexpr int TOTK = S + N;          // 2064 key rows per batch
constexpr int KPAD = 2080;           // padded to multiple of 32
constexpr int QT   = TOTQ / 16;      // 65 q-tiles of 16

constexpr size_t SEQ_OUT_OFF  = 0;
constexpr size_t SEQ_MASK_OFF = (size_t)B * NSL * D;                  // 4194304
constexpr size_t NS_OUT_OFF   = SEQ_MASK_OFF + (size_t)B * NSL;       // 4198400
constexpr size_t NS_MASK_OFF  = NS_OUT_OFF + (size_t)B * N * D;       // 4263936

typedef __attribute__((ext_vector_type(16))) __bf16 v16bf;
typedef __attribute__((ext_vector_type(8)))  float  v8f;
using u16 = unsigned short;
using u8  = unsigned char;

union Frag {
  uint4 q[2];
  v16bf v;
  u16   u[16];
};

__device__ __forceinline__ u16 f2bf(float f) {
  union { float f; unsigned int u; } c; c.f = f;
  unsigned int u = c.u;
  u += 0x7FFFu + ((u >> 16) & 1u);           // round-to-nearest-even
  return (u16)(u >> 16);
}

__device__ __forceinline__ v8f wmma_bf16(v16bf a, v16bf b, v8f c) {
  // D = A(16x32 bf16) * B(32x16 bf16) + C(16x16 f32)
  return __builtin_amdgcn_wmma_f32_16x16x32_bf16(false, a, false, b,
                                                 (short)0, c, false, false);
}

// CDNA5 async global->LDS copy (GV mode), tracked by ASYNCcnt.
// lds_addr: 32-bit LDS byte address; gaddr: 64-bit global address.
__device__ __forceinline__ void async_load_b128(unsigned lds_addr, const void* gaddr) {
  asm volatile("global_load_async_to_lds_b128 %0, %1, off"
               :: "v"(lds_addr), "v"(gaddr)
               : "memory");
}
__device__ __forceinline__ void async_wait0() {
  asm volatile("s_wait_asynccnt 0" ::: "memory");
}

// ---------------------------------------------------------------------------
// f32 -> bf16 conversion (weights)
// ---------------------------------------------------------------------------
__global__ void k_f32_to_bf16(const float* __restrict__ src,
                              u16* __restrict__ dst, int n) {
  int i = blockIdx.x * 256 + threadIdx.x;
  if (i < n) dst[i] = f2bf(src[i]);
}

// ---------------------------------------------------------------------------
// RMSNorm over rows of length D=1024; bf16 out + optional f32 out
// ---------------------------------------------------------------------------
__global__ void k_rmsnorm(const float* __restrict__ in, const float* __restrict__ w,
                          u16* __restrict__ outbf, float* __restrict__ outf, int rows) {
  int r = blockIdx.x;
  if (r >= rows) return;
  int t = threadIdx.x;
  const float* x = in + (size_t)r * D;
  float v[4]; float ss = 0.f;
  #pragma unroll
  for (int j = 0; j < 4; ++j) { v[j] = x[t + 256 * j]; ss += v[j] * v[j]; }
  #pragma unroll
  for (int o = 16; o >= 1; o >>= 1) ss += __shfl_xor(ss, o, 32);
  __shared__ float red[8];
  if ((t & 31) == 0) red[t >> 5] = ss;
  __syncthreads();
  float tot = 0.f;
  #pragma unroll
  for (int wv = 0; wv < 8; ++wv) tot += red[wv];
  const float rs = rsqrtf(tot * (1.0f / (float)D) + 1e-6f);
  #pragma unroll
  for (int j = 0; j < 4; ++j) {
    int d = t + 256 * j;
    float y = v[j] * rs * w[d];
    outbf[(size_t)r * D + d] = f2bf(y);
    if (outf) outf[(size_t)r * D + d] = y;
  }
}

// RMSNorm of the ns rows of resid (rows b*1040+1024+n), f32 out -> hns (64 x D)
__global__ void k_rmsnorm_ns(const float* __restrict__ resid, const float* __restrict__ w,
                             float* __restrict__ hns) {
  int r = blockIdx.x;              // 0..63
  int b = r >> 4, n = r & 15;
  int t = threadIdx.x;
  const float* x = resid + ((size_t)(b * TOTQ + NSL + n)) * D;
  float v[4]; float ss = 0.f;
  #pragma unroll
  for (int j = 0; j < 4; ++j) { v[j] = x[t + 256 * j]; ss += v[j] * v[j]; }
  #pragma unroll
  for (int o = 16; o >= 1; o >>= 1) ss += __shfl_xor(ss, o, 32);
  __shared__ float red[8];
  if ((t & 31) == 0) red[t >> 5] = ss;
  __syncthreads();
  float tot = 0.f;
  #pragma unroll
  for (int wv = 0; wv < 8; ++wv) tot += red[wv];
  const float rs = rsqrtf(tot * (1.0f / (float)D) + 1e-6f);
  #pragma unroll
  for (int j = 0; j < 4; ++j) {
    int d = t + 256 * j;
    hns[(size_t)r * D + d] = v[j] * rs * w[d];
  }
}

// ---------------------------------------------------------------------------
// Assemble XQ = [x_seq[:, -nsl:], x_ns] (bf16, B*1040 x D)
// ---------------------------------------------------------------------------
__global__ void k_assemble_xq(const u16* __restrict__ Xseq, const u16* __restrict__ Xns,
                              u16* __restrict__ XQ) {
  int i = blockIdx.x * 256 + threadIdx.x;
  if (i >= B * TOTQ * D) return;
  int r = i >> 10, d = i & 1023;
  int b = r / TOTQ, q = r % TOTQ;
  u16 v = (q < NSL) ? Xseq[((size_t)(b * S + NSL + q)) * D + d]
                    : Xns[((size_t)(b * N + (q - NSL))) * D + d];
  XQ[i] = v;
}

// ---------------------------------------------------------------------------
// key-validity mask (causal part handled in-attention) + K/Vt zero padding
// ---------------------------------------------------------------------------
__global__ void k_build_kmask(const u8* __restrict__ seqmask, const u8* __restrict__ nsmask,
                              u8* __restrict__ kmask) {
  int i = blockIdx.x * 256 + threadIdx.x;
  if (i >= B * KPAD) return;
  int b = i / KPAD, k = i % KPAD;
  u8 v = 0;
  if (k < S) v = seqmask[b * S + k];
  else if (k < TOTK) v = nsmask[b * N + (k - S)];
  kmask[i] = v;
}

__global__ void k_pad_kv(u16* __restrict__ Kb, u16* __restrict__ Vt) {
  int i = blockIdx.x * 256 + threadIdx.x;
  if (i < B * (KPAD - TOTK) * D) {
    int b = i / ((KPAD - TOTK) * D);
    int rem = i % ((KPAD - TOTK) * D);
    int j = rem / D, d = rem % D;
    Kb[((size_t)(b * KPAD + TOTK + j)) * D + d] = 0;
  }
  if (i < B * H * HD * (KPAD - TOTK)) {
    int rrow = i / (KPAD - TOTK), j = i % (KPAD - TOTK);
    Vt[(size_t)rrow * KPAD + TOTK + j] = 0;
  }
}

// ---------------------------------------------------------------------------
// Tiled bf16 WMMA GEMM: C[M x Ncols] = A[M x K] * W[K x Ncols] (+epilogue)
// Block = 256 threads (8 waves), 128x128 C tile, k-step 32.
// Double-buffered LDS; A tile staged with GLOBAL_LOAD_ASYNC_TO_LDS_B128
// (ASYNCcnt), W tile prefetched into VGPRs, both overlapping the WMMAs.
// EPI: 0 = bf16 out w/ per-batch row remap (inBatchRows -> outBatchStride)
//      1 = f32 out       2 = +bias, SiLU, bf16 out     3 = +bias, f32 out
//      4 = bf16 out to V-transposed layout [(b*H+h)*HD+dim][KPAD] at key=row
// ---------------------------------------------------------------------------
template <int EPI>
__global__ __launch_bounds__(256) void k_gemm(
    const u16* __restrict__ A, const u16* __restrict__ W,
    const float* __restrict__ bias, void* __restrict__ Cout,
    int M, int K, int Ncols, int inBatchRows, int outBatchStride) {
  constexpr int LDK  = 40;                   // padded LDS row (halves)
  constexpr int TILE = 128 * LDK;            // halves per buffer
  __shared__ u16 sA[2 * TILE];
  __shared__ u16 sB[2 * TILE];

  const int t = threadIdx.x;
  const int rowBase = blockIdx.y * 128;
  const int colBase = blockIdx.x * 128;
  const int wave = t >> 5, lane = t & 31;
  const int wm = wave & 3, wn = wave >> 2;   // wave tile: 32 rows x 64 cols
  const int half = lane >> 4, lr = lane & 15;
  const int sel8 = half * 8;

  // 32-bit LDS byte address of sA (generic pointer low bits = LDS offset)
  const unsigned ldsA_base = (unsigned)(uintptr_t)(void*)&sA[0];

  const v8f vzero = {0, 0, 0, 0, 0, 0, 0, 0};
  v8f acc[2][4];
  #pragma unroll
  for (int mt = 0; mt < 2; ++mt)
    #pragma unroll
    for (int nt = 0; nt < 4; ++nt) acc[mt][nt] = vzero;

  uint4 breg[2];

  // --- tile staging helpers -----------------------------------------------
  auto stageA_async = [&](int k0, int buf) {
    #pragma unroll
    for (int p = 0; p < 2; ++p) {
      int idx = p * 2048 + t * 8;
      int r = idx >> 5, kk = idx & 31;
      int gr = rowBase + r;
      if (gr >= M) gr = M - 1;               // clamp; rows discarded in epilogue
      const void* g = A + (size_t)gr * K + k0 + kk;
      unsigned l = ldsA_base + (unsigned)((buf * TILE + r * LDK + kk) * 2);
      async_load_b128(l, g);
    }
  };
  auto loadB = [&](int k0) {
    #pragma unroll
    for (int p = 0; p < 2; ++p) {
      int idx = p * 2048 + t * 8;
      int kr = idx >> 7, nc = idx & 127;
      breg[p] = *(const uint4*)(W + (size_t)(k0 + kr) * Ncols + colBase + nc);
    }
  };
  auto storeB = [&](int buf) {               // transpose so B-frags are contiguous
    #pragma unroll
    for (int p = 0; p < 2; ++p) {
      int idx = p * 2048 + t * 8;
      int kr = idx >> 7, nc = idx & 127;
      const u16* u = (const u16*)&breg[p];
      #pragma unroll
      for (int j = 0; j < 8; ++j) sB[buf * TILE + (nc + j) * LDK + kr] = u[j];
    }
  };

  // --- prologue: stage tile 0 ---------------------------------------------
  stageA_async(0, 0);
  loadB(0);
  storeB(0);
  async_wait0();
  __syncthreads();

  const int nk = K / 32;
  for (int kt = 0; kt < nk; ++kt) {
    const int cur = kt & 1;
    const bool more = (kt + 1) < nk;
    if (more) {
      stageA_async((kt + 1) * 32, cur ^ 1);  // async, overlaps WMMAs below
      loadB((kt + 1) * 32);                  // in flight during WMMAs
    }

    // --- compute on current buffer ---
    const u16* bA = &sA[cur * TILE];
    const u16* bB = &sB[cur * TILE];
    Frag fa[2], fb[4];
    #pragma unroll
    for (int mt = 0; mt < 2; ++mt) {
      const int mr = wm * 32 + mt * 16 + lr;
      fa[mt].q[0] = *(const uint4*)&bA[mr * LDK + sel8];
      fa[mt].q[1] = *(const uint4*)&bA[mr * LDK + 16 + sel8];
    }
    #pragma unroll
    for (int nt = 0; nt < 4; ++nt) {
      const int nc = wn * 64 + nt * 16 + lr;
      fb[nt].q[0] = *(const uint4*)&bB[nc * LDK + sel8];
      fb[nt].q[1] = *(const uint4*)&bB[nc * LDK + 16 + sel8];
    }
    #pragma unroll
    for (int mt = 0; mt < 2; ++mt)
      #pragma unroll
      for (int nt = 0; nt < 4; ++nt)
        acc[mt][nt] = wmma_bf16(fa[mt].v, fb[nt].v, acc[mt][nt]);

    if (more) storeB(cur ^ 1);
    async_wait0();
    __syncthreads();
  }

  // --- epilogue ---
  #pragma unroll
  for (int mt = 0; mt < 2; ++mt) {
    #pragma unroll
    for (int nt = 0; nt < 4; ++nt) {
      #pragma unroll
      for (int i = 0; i < 8; ++i) {
        const int row = rowBase + wm * 32 + mt * 16 + i + 8 * half;
        if (row >= M) continue;
        const int col = colBase + wn * 64 + nt * 16 + lr;
        float v = acc[mt][nt][i];
        if constexpr (EPI == 0) {
          const int bidx = row / inBatchRows, srow = row % inBatchRows;
          ((u16*)Cout)[((size_t)(bidx * outBatchStride + srow)) * Ncols + col] = f2bf(v);
        } else if constexpr (EPI == 1) {
          ((float*)Cout)[(size_t)row * Ncols + col] = v;
        } else if constexpr (EPI == 2) {
          v += bias[col];
          v = v / (1.f + __expf(-v));
          ((u16*)Cout)[(size_t)row * Ncols + col] = f2bf(v);
        } else if constexpr (EPI == 3) {
          v += bias[col];
          ((float*)Cout)[(size_t)row * Ncols + col] = v;
        } else {  // EPI == 4 : V transposed [(b*H+h)*HD+dim][KPAD]
          const int bidx = row / inBatchRows, srow = row % inBatchRows;
          const int hh = col >> 6, dim = col & 63;
          ((u16*)Cout)[((size_t)((bidx * H + hh) * HD + dim)) * KPAD + srow] = f2bf(v);
        }
      }
    }
  }
}

// ---------------------------------------------------------------------------
// ns (expert) QKV: q/k/v[b,n,:] = x_ns[b,n,:] @ W[n]  (f32, bandwidth-bound)
// Writes directly into the padded Q / K / Vt buffers at the ns rows.
// ---------------------------------------------------------------------------
__global__ __launch_bounds__(256) void k_ns_qkv(
    const float* __restrict__ xns, const float* __restrict__ Wqn,
    const float* __restrict__ Wkn, const float* __restrict__ Wvn,
    u16* __restrict__ Q, u16* __restrict__ Kb, u16* __restrict__ Vt) {
  const int n = blockIdx.x;
  const int o = blockIdx.y * 256 + threadIdx.x;
  __shared__ float sx[4][D];
  for (int j = 0; j < 16; ++j) {
    int idx = threadIdx.x + j * 256;
    int b = idx >> 10, d = idx & 1023;
    sx[b][d] = xns[((size_t)(b * N + n)) * D + d];
  }
  __syncthreads();
  float aq[4] = {0, 0, 0, 0}, ak[4] = {0, 0, 0, 0}, av[4] = {0, 0, 0, 0};
  const size_t wbase = (size_t)n * D * D;
  for (int d = 0; d < D; ++d) {
    const float wq = Wqn[wbase + (size_t)d * D + o];
    const float wk = Wkn[wbase + (size_t)d * D + o];
    const float wv = Wvn[wbase + (size_t)d * D + o];
    #pragma unroll
    for (int b = 0; b < 4; ++b) {
      aq[b] += sx[b][d] * wq;
      ak[b] += sx[b][d] * wk;
      av[b] += sx[b][d] * wv;
    }
  }
  const int hh = o >> 6, dim = o & 63;
  #pragma unroll
  for (int b = 0; b < 4; ++b) {
    Q[((size_t)(b * TOTQ + NSL + n)) * D + o] = f2bf(aq[b]);
    Kb[((size_t)(b * KPAD + S + n)) * D + o] = f2bf(ak[b]);
    Vt[((size_t)((b * H + hh) * HD + dim)) * KPAD + S + n] = f2bf(av[b]);
  }
}

// ---------------------------------------------------------------------------
// Flash attention: one wave per (b, h, 16-query tile). bf16 WMMA.
// ---------------------------------------------------------------------------
__global__ __launch_bounds__(32) void k_attention(
    const u16* __restrict__ Q, const u16* __restrict__ Kb,
    const u16* __restrict__ Vt, const u8* __restrict__ kmask,
    u16* __restrict__ att) {
  const int bid = blockIdx.x;
  const int qt = bid % QT;
  const int h  = (bid / QT) & (H - 1);
  const int b  = bid / (QT * H);
  const int lane = threadIdx.x;
  const int half = lane >> 4, lr = lane & 15;
  const int sel8 = half * 8;
  const float scale = 0.125f;  // 1/sqrt(64)

  Frag qa0, qa1;  // Q fragments for head-dim 0..31 and 32..63
  {
    const u16* qp = Q + ((size_t)(b * TOTQ + qt * 16 + lr)) * D + h * HD;
    qa0.q[0] = *(const uint4*)(qp + sel8);
    qa0.q[1] = *(const uint4*)(qp + 16 + sel8);
    qa1.q[0] = *(const uint4*)(qp + 32 + sel8);
    qa1.q[1] = *(const uint4*)(qp + 48 + sel8);
  }

  const v8f vzero = {0, 0, 0, 0, 0, 0, 0, 0};
  float m[8], lsum[8];
  v8f o[4];
  #pragma unroll
  for (int i = 0; i < 8; ++i) { m[i] = -1e30f; lsum[i] = 0.f; }
  #pragma unroll
  for (int dt = 0; dt < 4; ++dt) o[dt] = vzero;

  __shared__ u16 sP[16 * 40];

  for (int kt = 0; kt < KPAD / 32; ++kt) {
    const int key0 = kt * 32;

    // S = Q * K^T (16q x 32keys), K^T fragments loaded directly from K rows
    v8f s[2];
    #pragma unroll
    for (int ct = 0; ct < 2; ++ct) {
      v8f sc = vzero;
      const int key = key0 + ct * 16 + lr;
      const u16* kp = Kb + ((size_t)(b * KPAD + key)) * D + h * HD;
      Frag fk;
      fk.q[0] = *(const uint4*)(kp + sel8);
      fk.q[1] = *(const uint4*)(kp + 16 + sel8);
      sc = wmma_bf16(qa0.v, fk.v, sc);
      fk.q[0] = *(const uint4*)(kp + 32 + sel8);
      fk.q[1] = *(const uint4*)(kp + 48 + sel8);
      sc = wmma_bf16(qa1.v, fk.v, sc);
      s[ct] = sc;
    }

    // online softmax update
    const int keyA = key0 + lr;
    const int keyB = key0 + 16 + lr;
    const bool mA = kmask[b * KPAD + keyA] != 0;
    const bool mB = kmask[b * KPAD + keyB] != 0;
    float p0[8], p1[8], corr[8];
    #pragma unroll
    for (int i = 0; i < 8; ++i) {
      const int qpos = NSL + qt * 16 + i + 8 * half;  // global position of q row
      float l0 = (mA && keyA <= qpos) ? s[0][i] * scale : -10000.f;
      float l1 = (mB && keyB <= qpos) ? s[1][i] * scale : -10000.f;
      float mx = fmaxf(l0, l1);
      #pragma unroll
      for (int off = 8; off >= 1; off >>= 1) mx = fmaxf(mx, __shfl_xor(mx, off, 32));
      const float mn = fmaxf(m[i], mx);
      corr[i] = __expf(m[i] - mn);
      m[i] = mn;
      const float e0 = __expf(l0 - mn);
      const float e1 = __expf(l1 - mn);
      float ps = e0 + e1;
      #pragma unroll
      for (int off = 8; off >= 1; off >>= 1) ps += __shfl_xor(ps, off, 32);
      lsum[i] = lsum[i] * corr[i] + ps;
      p0[i] = e0; p1[i] = e1;
    }
    #pragma unroll
    for (int dt = 0; dt < 4; ++dt)
      #pragma unroll
      for (int i = 0; i < 8; ++i) o[dt][i] = o[dt][i] * corr[i];

    // C-layout P -> A-layout fragment via LDS
    __syncthreads();
    #pragma unroll
    for (int i = 0; i < 8; ++i) {
      sP[(i + 8 * half) * 40 + lr]      = f2bf(p0[i]);
      sP[(i + 8 * half) * 40 + 16 + lr] = f2bf(p1[i]);
    }
    __syncthreads();
    Frag pf;
    pf.q[0] = *(const uint4*)&sP[lr * 40 + sel8];
    pf.q[1] = *(const uint4*)&sP[lr * 40 + 16 + sel8];

    // O += P * V ; V fragments contiguous in keys thanks to Vt layout
    #pragma unroll
    for (int dt = 0; dt < 4; ++dt) {
      const int dim = dt * 16 + lr;
      const u16* vp = Vt + ((size_t)((b * H + h) * HD + dim)) * KPAD + key0;
      Frag fv;
      fv.q[0] = *(const uint4*)(vp + sel8);
      fv.q[1] = *(const uint4*)(vp + 16 + sel8);
      o[dt] = wmma_bf16(pf.v, fv.v, o[dt]);
    }
  }

  #pragma unroll
  for (int dt = 0; dt < 4; ++dt)
    #pragma unroll
    for (int i = 0; i < 8; ++i) {
      const int row = qt * 16 + i + 8 * half;
      const float val = o[dt][i] / lsum[i];
      att[((size_t)(b * TOTQ + row)) * D + h * HD + dt * 16 + lr] = f2bf(val);
    }
}

// ---------------------------------------------------------------------------
// residual: resid = base_tokens + att@Wo * mask
// ---------------------------------------------------------------------------
__global__ void k_residual(const float* __restrict__ seqtok, const float* __restrict__ nstok,
                           const u8* __restrict__ seqmask, const u8* __restrict__ nsmask,
                           const float* __restrict__ attWo, float* __restrict__ resid) {
  int i = blockIdx.x * 256 + threadIdx.x;
  if (i >= B * TOTQ * D) return;
  int r = i >> 10, d = i & 1023;
  int b = r / TOTQ, q = r % TOTQ;
  float base, msk;
  if (q < NSL) {
    base = seqtok[((size_t)(b * S + NSL + q)) * D + d];
    msk = seqmask[b * S + NSL + q] ? 1.f : 0.f;
  } else {
    int n = q - NSL;
    base = nstok[((size_t)(b * N + n)) * D + d];
    msk = nsmask[b * N + n] ? 1.f : 0.f;
  }
  resid[i] = base + attWo[i] * msk;
}

// ---------------------------------------------------------------------------
// expert FFN (f32, bandwidth-bound): up+SiLU then down (+bias) over 4 rows
// ---------------------------------------------------------------------------
__global__ __launch_bounds__(256) void k_expert_up(
    const float* __restrict__ hns, const float* __restrict__ Wu,
    const float* __restrict__ bu, float* __restrict__ nsmid) {
  const int n = blockIdx.x;
  const int f = blockIdx.y * 256 + threadIdx.x;
  __shared__ float sx[4][D];
  for (int j = 0; j < 16; ++j) {
    int idx = threadIdx.x + j * 256;
    int b = idx >> 10, d = idx & 1023;
    sx[b][d] = hns[((size_t)(b * N + n)) * D + d];
  }
  __syncthreads();
  float acc[4] = {0, 0, 0, 0};
  const size_t wbase = (size_t)n * D * F;
  for (int d = 0; d < D; ++d) {
    const float wv = Wu[wbase + (size_t)d * F + f];
    #pragma unroll
    for (int b = 0; b < 4; ++b) acc[b] += sx[b][d] * wv;
  }
  const float bb = bu[n * F + f];
  #pragma unroll
  for (int b = 0; b < 4; ++b) {
    float v = acc[b] + bb;
    v = v / (1.f + __expf(-v));
    nsmid[((size_t)(b * N + n)) * F + f] = v;
  }
}

__global__ __launch_bounds__(256) void k_expert_down(
    const float* __restrict__ nsmid, const float* __restrict__ Wd,
    const float* __restrict__ bd, float* __restrict__ ffn) {
  const int n = blockIdx.x;
  const int d = blockIdx.y * 256 + threadIdx.x;
  __shared__ float sy[4][2048];
  float acc[4] = {0, 0, 0, 0};
  const size_t wbase = (size_t)n * F * D;
  for (int c = 0; c < 2; ++c) {
    __syncthreads();
    for (int j = 0; j < 32; ++j) {
      int idx = threadIdx.x + j * 256;
      int b = idx >> 11, f = idx & 2047;
      sy[b][f] = nsmid[((size_t)(b * N + n)) * F + c * 2048 + f];
    }
    __syncthreads();
    for (int f = 0; f < 2048; ++f) {
      const float wv = Wd[wbase + (size_t)(c * 2048 + f) * D + d];
      #pragma unroll
      for (int b = 0; b < 4; ++b) acc[b] += sy[b][f] * wv;
    }
  }
  const float bb = bd[n * D + d];
  #pragma unroll
  for (int b = 0; b < 4; ++b)
    ffn[((size_t)(b * TOTQ + NSL + n)) * D + d] = acc[b] + bb;
}

// ---------------------------------------------------------------------------
// final: out = (resid + ffn*mask) * mask, scattered to seq_out / ns_out
// ---------------------------------------------------------------------------
__global__ void k_final(const float* __restrict__ resid, const float* __restrict__ ffn,
                        const u8* __restrict__ seqmask, const u8* __restrict__ nsmask,
                        float* __restrict__ out) {
  int i = blockIdx.x * 256 + threadIdx.x;
  if (i >= B * TOTQ * D) return;
  int r = i >> 10, d = i & 1023;
  int b = r / TOTQ, q = r % TOTQ;
  float msk; size_t oidx;
  if (q < NSL) {
    msk = seqmask[b * S + NSL + q] ? 1.f : 0.f;
    oidx = SEQ_OUT_OFF + ((size_t)(b * NSL + q)) * D + d;
  } else {
    int n = q - NSL;
    msk = nsmask[b * N + n] ? 1.f : 0.f;
    oidx = NS_OUT_OFF + ((size_t)(b * N + n)) * D + d;
  }
  out[oidx] = (resid[i] + ffn[i] * msk) * msk;
}

__global__ void k_masks(const u8* __restrict__ seqmask, const u8* __restrict__ nsmask,
                        float* __restrict__ out) {
  int i = blockIdx.x * 256 + threadIdx.x;
  if (i < B * NSL) {
    int b = i >> 10, q = i & 1023;
    out[SEQ_MASK_OFF + i] = seqmask[b * S + NSL + q] ? 1.f : 0.f;
  } else if (i < B * NSL + B * N) {
    int j = i - B * NSL;
    out[NS_MASK_OFF + j] = nsmask[j] ? 1.f : 0.f;
  }
}

// ---------------------------------------------------------------------------
// host launcher
// ---------------------------------------------------------------------------
extern "C" void kernel_launch(void* const* d_in, const int* in_sizes, int n_in,
                              void* d_out, int out_size, void* d_ws, size_t ws_size,
                              hipStream_t stream) {
  const float* seq_tok   = (const float*)d_in[0];
  const u8*    seq_mask  = (const u8*)d_in[1];
  const float* ns_tok    = (const float*)d_in[2];
  const u8*    ns_mask   = (const u8*)d_in[3];
  const float* attn_w    = (const float*)d_in[5];
  const float* ffn_w     = (const float*)d_in[6];
  const float* Wq        = (const float*)d_in[7];
  const float* Wk        = (const float*)d_in[8];
  const float* Wv        = (const float*)d_in[9];
  const float* Wo        = (const float*)d_in[10];
  const float* ns_q_w    = (const float*)d_in[11];
  const float* ns_k_w    = (const float*)d_in[12];
  const float* ns_v_w    = (const float*)d_in[13];
  const float* seq_up_w  = (const float*)d_in[14];
  const float* seq_up_b  = (const float*)d_in[15];
  const float* seq_dn_w  = (const float*)d_in[16];
  const float* seq_dn_b  = (const float*)d_in[17];
  const float* ns_up_w   = (const float*)d_in[18];
  const float* ns_up_b   = (const float*)d_in[19];
  const float* ns_dn_w   = (const float*)d_in[20];
  const float* ns_dn_b   = (const float*)d_in[21];
  float* out = (float*)d_out;

  char* ws = (char*)d_ws;
  size_t off = 0;
  auto alloc = [&](size_t bytes) -> char* {
    char* p = ws + off;
    off += (bytes + 255) & ~(size_t)255;
    return p;
  };
  u16* Wq_c  = (u16*)alloc((size_t)D * D * 2);
  u16* Wk_c  = (u16*)alloc((size_t)D * D * 2);
  u16* Wv_c  = (u16*)alloc((size_t)D * D * 2);
  u16* Wo_c  = (u16*)alloc((size_t)D * D * 2);
  u16* up_c  = (u16*)alloc((size_t)D * F * 2);
  u16* dn_c  = (u16*)alloc((size_t)F * D * 2);
  u16* Xseq  = (u16*)alloc((size_t)B * S * D * 2);
  u16* Xnsb  = (u16*)alloc((size_t)B * N * D * 2);
  float* Xnsf = (float*)alloc((size_t)B * N * D * 4);
  u16* XQ    = (u16*)alloc((size_t)B * TOTQ * D * 2);
  u16* Qb    = (u16*)alloc((size_t)B * TOTQ * D * 2);
  u16* Kb    = (u16*)alloc((size_t)B * KPAD * D * 2);
  u16* Vt    = (u16*)alloc((size_t)B * KPAD * D * 2);
  u8*  kmask = (u8*)alloc((size_t)B * KPAD);
  u16* attb  = (u16*)alloc((size_t)B * TOTQ * D * 2);
  float* attWo = (float*)alloc((size_t)B * TOTQ * D * 4);
  float* resid = (float*)alloc((size_t)B * TOTQ * D * 4);
  u16* hbuf  = (u16*)alloc((size_t)B * TOTQ * D * 2);
  float* hns = (float*)alloc((size_t)B * N * D * 4);
  u16* mid   = (u16*)alloc((size_t)B * TOTQ * F * 2);
  float* nsmid = (float*)alloc((size_t)B * N * F * 4);
  float* ffn = (float*)alloc((size_t)B * TOTQ * D * 4);
  (void)ws_size; (void)in_sizes; (void)n_in; (void)out_size;

  auto cvt = [&](const float* s, u16* d, size_t n) {
    k_f32_to_bf16<<<(unsigned)((n + 255) / 256), 256, 0, stream>>>(s, d, (int)n);
  };
  cvt(Wq, Wq_c, (size_t)D * D);
  cvt(Wk, Wk_c, (size_t)D * D);
  cvt(Wv, Wv_c, (size_t)D * D);
  cvt(Wo, Wo_c, (size_t)D * D);
  cvt(seq_up_w, up_c, (size_t)D * F);
  cvt(seq_dn_w, dn_c, (size_t)F * D);

  k_build_kmask<<<(B * KPAD + 255) / 256, 256, 0, stream>>>(seq_mask, ns_mask, kmask);
  k_pad_kv<<<(B * H * HD * (KPAD - TOTK) + 255) / 256, 256, 0, stream>>>(Kb, Vt);

  k_rmsnorm<<<B * S, 256, 0, stream>>>(seq_tok, attn_w, Xseq, nullptr, B * S);
  k_rmsnorm<<<B * N, 256, 0, stream>>>(ns_tok, attn_w, Xnsb, Xnsf, B * N);
  k_assemble_xq<<<(B * TOTQ * D + 255) / 256, 256, 0, stream>>>(Xseq, Xnsb, XQ);

  // QKV projections (WMMA, async-LDS pipelined)
  k_gemm<0><<<dim3(D / 128, (B * TOTQ + 127) / 128), 256, 0, stream>>>(
      XQ, Wq_c, nullptr, Qb, B * TOTQ, D, D, B * TOTQ, 0);
  k_gemm<0><<<dim3(D / 128, (B * S) / 128), 256, 0, stream>>>(
      Xseq, Wk_c, nullptr, Kb, B * S, D, D, S, KPAD);
  k_gemm<4><<<dim3(D / 128, (B * S) / 128), 256, 0, stream>>>(
      Xseq, Wv_c, nullptr, Vt, B * S, D, D, S, 0);
  k_ns_qkv<<<dim3(N, D / 256), 256, 0, stream>>>(Xnsf, ns_q_w, ns_k_w, ns_v_w, Qb, Kb, Vt);

  // attention (WMMA flash)
  k_attention<<<B * H * QT, 32, 0, stream>>>(Qb, Kb, Vt, kmask, attb);

  // output projection + residual
  k_gemm<1><<<dim3(D / 128, (B * TOTQ + 127) / 128), 256, 0, stream>>>(
      attb, Wo_c, nullptr, attWo, B * TOTQ, D, D, B * TOTQ, 0);
  k_residual<<<(B * TOTQ * D + 255) / 256, 256, 0, stream>>>(
      seq_tok, ns_tok, seq_mask, ns_mask, attWo, resid);

  // FFN
  k_rmsnorm<<<B * TOTQ, 256, 0, stream>>>(resid, ffn_w, hbuf, nullptr, B * TOTQ);
  k_gemm<2><<<dim3(F / 128, (B * TOTQ + 127) / 128), 256, 0, stream>>>(
      hbuf, up_c, seq_up_b, mid, B * TOTQ, D, F, B * TOTQ, 0);
  k_gemm<3><<<dim3(D / 128, (B * TOTQ + 127) / 128), 256, 0, stream>>>(
      mid, dn_c, seq_dn_b, ffn, B * TOTQ, F, D, B * TOTQ, 0);

  // expert FFN overwrites the ns rows of ffn
  k_rmsnorm_ns<<<B * N, 256, 0, stream>>>(resid, ffn_w, hns);
  k_expert_up<<<dim3(N, F / 256), 256, 0, stream>>>(hns, ns_up_w, ns_up_b, nsmid);
  k_expert_down<<<dim3(N, D / 256), 256, 0, stream>>>(nsmid, ns_dn_w, ns_dn_b, ffn);

  // outputs
  k_final<<<(B * TOTQ * D + 255) / 256, 256, 0, stream>>>(resid, ffn, seq_mask, ns_mask, out);
  k_masks<<<(B * NSL + B * N + 255) / 256, 256, 0, stream>>>(seq_mask, ns_mask, out);
}
